// RecSAE_38646115729649
// MI455X (gfx1250) — matrix operand order repaired and checked by reading, not compiled, not run on hardware
//
#include <hip/hip_runtime.h>

typedef __attribute__((ext_vector_type(16))) _Float16 v16h;
typedef __attribute__((ext_vector_type(8)))  float    v8f;
typedef __attribute__((ext_vector_type(4)))  float    v4f;

#define B_ROWS   16384
#define D_DIM    256
#define L_DIM    4096
#define ROWS_WG  16
#define KBLK     8      // 256 / 32
#define NTILES   256    // 4096 / 16
#define TPW      32     // tiles per wave (8 waves)

// ---------------------------------------------------------------------------
// Kernel 1: convert W_enc (fp32 [256,4096] row-major) into f16 fragments laid
// out as V_WMMA_F32_16X16X32_F16 B operands in contiguous k order (as A), packed
// per-lane-contiguous (32 B per lane) so the GEMM loads 2x global_load_b128.
// B-layout (32x16, K x N): lane<16  -> N=lane,    K = h      (h = half 0..15)
//                          lane>=16 -> N=lane-16, K = 16+h
// Tile order in ws: tile = nt*8 + kb ; element = tile*32 + lane (each v16h).
// ---------------------------------------------------------------------------
__global__ void __launch_bounds__(256)
prep_wenc_f16(const float* __restrict__ W_enc, v16h* __restrict__ Bt) {
    int gid  = blockIdx.x * 256 + threadIdx.x;   // 65536 threads total
    int lane = gid & 31;
    int tile = gid >> 5;                          // 0..2047
    int nt   = tile >> 3;
    int kb   = tile & 7;
    int n    = lane & 15;
    int kofs = (lane & 16) ? 16 : 0;
    v16h b;
#pragma unroll
    for (int h = 0; h < 16; ++h) {
        int k = kb * 32 + kofs + h;
        b[h] = (_Float16)W_enc[(size_t)k * L_DIM + nt * 16 + n];
    }
    Bt[(size_t)tile * 32 + lane] = b;
}

// ---------------------------------------------------------------------------
// Kernel 2: fused encode (WMMA) -> top-8 -> dense acts -> sparse decode.
// One workgroup = 16 rows of x. pre_acts stripe (16x4096 f32 = 256KB) lives
// entirely in LDS (CDNA5: 320KB/WGP).
// ---------------------------------------------------------------------------
__global__ void __launch_bounds__(256)
sae_fused(const float* __restrict__ x, const float* __restrict__ b_pre,
          const float* __restrict__ b_enc, const float* __restrict__ W_dec,
          const v16h* __restrict__ Bt,
          float* __restrict__ acts, float* __restrict__ recon) {
    extern __shared__ char smem[];
    v16h*  sA    = (v16h*)smem;                          // 256 * 32B  = 8192
    float* spre  = (float*)(smem + 8192);                // 16*4096*4 = 262144
    float* candV = (float*)(smem + 8192 + 262144);       // 16*64*4   = 4096
    int*   candI = (int*)  (smem + 8192 + 262144 + 4096);// 4096
    float* finV  = (float*)(smem + 8192 + 262144 + 8192);// 16*8*4    = 512
    int*   finI  = (int*)  (smem + 8192 + 262144 + 8192 + 512); // 512

    const int tid  = threadIdx.x;
    const int lane = tid & 31;
    const int wave = tid >> 5;
    const int row0 = blockIdx.x * ROWS_WG;

    // ---- Stage A = (x - b_pre) as f16 into LDS in contiguous-k WMMA A order ----------
    // A 16x32: lane<16  -> M=lane,    K = h
    //          lane>=16 -> M=lane-16, K = 16+h
    {
        int kb   = tid >> 5;           // this thread fills k-block kb, lane l
        int l    = tid & 31;
        int m    = l & 15;
        int kofs = (l & 16) ? 16 : 0;
        const float* xr = x + (size_t)(row0 + m) * D_DIM;
        v16h a;
#pragma unroll
        for (int h = 0; h < 16; ++h) {
            int k = kb * 32 + h + kofs;
            a[h] = (_Float16)(xr[k] - b_pre[k]);
        }
        sA[kb * 32 + l] = a;
    }
    __syncthreads();

    // ---- Encoder GEMM: each wave owns a 512-wide N stripe (32 tiles), ----
    // ---- processed 2 tiles at a time -> 2 independent WMMA dep-chains ----
    v16h afrag[KBLK];
#pragma unroll
    for (int kb = 0; kb < KBLK; ++kb) afrag[kb] = sA[kb * 32 + lane];

    for (int t = 0; t < TPW; t += 2) {
        int nt0 = wave * TPW + t;
        const v16h* bp0 = Bt + (size_t)nt0 * (KBLK * 32) + lane;
        const v16h* bp1 = bp0 + (KBLK * 32);
        if (t + 2 < TPW)
            __builtin_prefetch((const void*)(bp0 + 2 * KBLK * 32), 0, 1);
        v8f acc0 = {0.f, 0.f, 0.f, 0.f, 0.f, 0.f, 0.f, 0.f};
        v8f acc1 = {0.f, 0.f, 0.f, 0.f, 0.f, 0.f, 0.f, 0.f};
#pragma unroll
        for (int kb = 0; kb < KBLK; ++kb) {
            v16h b0 = bp0[kb * 32];
            v16h b1 = bp1[kb * 32];
            acc0 = __builtin_amdgcn_wmma_f32_16x16x32_f16(
                false, afrag[kb], false, b0, (short)0, acc0, false, false);
            acc1 = __builtin_amdgcn_wmma_f32_16x16x32_f16(
                false, afrag[kb], false, b1, (short)0, acc1, false, false);
        }
        // C layout: vgpr r, lane<16 -> (M=r, N=lane); lane>=16 -> (M=r+8)
        int col0  = nt0 * 16 + (lane & 15);
        float be0 = b_enc[col0];
        float be1 = b_enc[col0 + 16];
        int mbase = (lane & 16) ? 8 : 0;
#pragma unroll
        for (int r = 0; r < 8; ++r) {
            spre[(mbase + r) * L_DIM + col0]      = acc0[r] + be0;
            spre[(mbase + r) * L_DIM + col0 + 16] = acc1[r] + be1;
        }
    }
    __syncthreads();

    // ---- Per-segment top-8: 8 threads/row, 512 values each (v4 loads) ----
    if (tid < 128) {
        int row = tid >> 3, seg = tid & 7;
        const v4f* rp4 = (const v4f*)(spre + row * L_DIM + seg * 512);
        float tv[8]; int ti[8];
#pragma unroll
        for (int q = 0; q < 8; ++q) { tv[q] = -3.4e38f; ti[q] = 0x7FFFFFFF; }
        for (int j = 0; j < 128; ++j) {
            v4f v4 = rp4[j];
#pragma unroll
            for (int c = 0; c < 4; ++c) {
                float v = v4[c];
                if (v > tv[7]) {          // strict > keeps lowest index on ties
                    tv[7] = v; ti[7] = seg * 512 + j * 4 + c;
#pragma unroll
                    for (int q = 7; q > 0; --q) {
                        if (tv[q] > tv[q - 1]) {
                            float fv = tv[q]; tv[q] = tv[q - 1]; tv[q - 1] = fv;
                            int   fi = ti[q]; ti[q] = ti[q - 1]; ti[q - 1] = fi;
                        }
                    }
                }
            }
        }
#pragma unroll
        for (int q = 0; q < 8; ++q) {
            candV[row * 64 + seg * 8 + q] = tv[q];
            candI[row * 64 + seg * 8 + q] = ti[q];
        }
    }
    __syncthreads();

    // ---- Merge 64 candidates -> global top-8 (lowest-index tie-break) ----
    if (tid < 16) {
        int row = tid;
        unsigned long long used = 0ull;
        for (int k = 0; k < 8; ++k) {
            float best = -3.4e38f; int bi = 0x7FFFFFFF; int bs = 0;
            for (int s = 0; s < 64; ++s) {
                if ((used >> s) & 1ull) continue;
                float v = candV[row * 64 + s];
                int  ii = candI[row * 64 + s];
                if (v > best || (v == best && ii < bi)) { best = v; bi = ii; bs = s; }
            }
            used |= (1ull << bs);
            finV[row * 8 + k] = best > 0.f ? best : 0.f;   // relu at scatter
            finI[row * 8 + k] = bi;
        }
    }
    __syncthreads();

    // ---- Rebuild stripe as sparse acts in LDS --------------------------
    {
        v4f z = {0.f, 0.f, 0.f, 0.f};
        v4f* sp4 = (v4f*)spre;
        for (int i = tid; i < (ROWS_WG * L_DIM) / 4; i += 256) sp4[i] = z;
    }
    __syncthreads();
    if (tid < 128) {
        int row = tid >> 3, k = tid & 7;
        spre[row * L_DIM + finI[row * 8 + k]] = finV[row * 8 + k];
    }
    __syncthreads();

    // ---- Stream acts stripe to HBM via async LDS->memory DMA -----------
    // (GLOBAL_STORE_ASYNC_FROM_LDS_B128: no VGPR staging; ASYNCcnt-tracked.
    //  Low 32 bits of a generic pointer to LDS are the LDS byte address.)
    {
        const v4f* sp4 = (const v4f*)spre;
        v4f* gout = (v4f*)(acts + (size_t)row0 * L_DIM);
        for (int i = tid; i < (ROWS_WG * L_DIM) / 4; i += 256) {
            unsigned long long ga = (unsigned long long)(uintptr_t)(gout + i);
            unsigned           la = (unsigned)(uintptr_t)(sp4 + i);
            asm volatile("global_store_async_from_lds_b128 %0, %1, off"
                         :: "v"(ga), "v"(la) : "memory");
        }
    }

    // ---- Sparse decode: recon = acts @ W_dec + b_pre (8 rows gathered) ---
    {
        float bp = b_pre[tid];               // blockDim.x == D_DIM == 256
        for (int r = 0; r < ROWS_WG; ++r) {
            float acc = bp;
#pragma unroll
            for (int k = 0; k < 8; ++k)
                acc += finV[r * 8 + k] *
                       W_dec[(size_t)finI[r * 8 + k] * D_DIM + tid];
            recon[(size_t)(row0 + r) * D_DIM + tid] = acc;
        }
    }

    // Ensure all async LDS->memory transfers have completed before the
    // workgroup (and its LDS allocation) retires.
    asm volatile("s_wait_asynccnt 0x0" ::: "memory");
}

// ---------------------------------------------------------------------------
extern "C" void kernel_launch(void* const* d_in, const int* in_sizes, int n_in,
                              void* d_out, int out_size, void* d_ws, size_t ws_size,
                              hipStream_t stream) {
    const float* x     = (const float*)d_in[0];
    const float* b_pre = (const float*)d_in[1];
    const float* W_enc = (const float*)d_in[2];
    const float* b_enc = (const float*)d_in[3];
    const float* W_dec = (const float*)d_in[4];

    float* acts  = (float*)d_out;                          // [16384, 4096]
    float* recon = acts + (size_t)B_ROWS * L_DIM;          // [16384, 256]

    v16h* Bt = (v16h*)d_ws;                                // 2 MB f16 W_enc

    prep_wenc_f16<<<256, 256, 0, stream>>>(W_enc, Bt);

    size_t lds = 8192 + 262144 + 4096 + 4096 + 512 + 512;  // 279,552 B < 320 KB
    (void)hipFuncSetAttribute((const void*)sae_fused,
                              hipFuncAttributeMaxDynamicSharedMemorySize,
                              (int)lds);
    sae_fused<<<B_ROWS / ROWS_WG, 256, lds, stream>>>(
        x, b_pre, b_enc, W_dec, Bt, acts, recon);
}
